// Attention_19250043420766
// MI455X (gfx1250) — matrix operand order repaired and checked by reading, not compiled, not run on hardware
//
#include <hip/hip_runtime.h>
#include <hip/hip_bf16.h>
#include <math.h>

typedef unsigned short u16;
typedef __attribute__((ext_vector_type(16))) __bf16 v16bf;
typedef __attribute__((ext_vector_type(8)))  float  v8f;

#define BM 64
#define BN 64
#define BK 32

// bf16 helpers (RNE)
__device__ __forceinline__ u16 f32_to_bf16(float f) {
    unsigned int u = __float_as_uint(f);
    u += 0x7FFFu + ((u >> 16) & 1u);
    return (u16)(u >> 16);
}
__device__ __forceinline__ float bf16_to_f32(u16 h) {
    return __uint_as_float(((unsigned int)h) << 16);
}

// ---- async copy global->LDS (gfx1250 ASYNCcnt path) with safe fallback ----
#if defined(__has_builtin)
#if __has_builtin(__builtin_amdgcn_global_load_async_to_lds_b128)
#define HAVE_ASYNC_LDS 1
#endif
#endif

#ifdef HAVE_ASYNC_LDS
// builtin expects clang-native vector pointers: 'vector_size(16) int *'
typedef __attribute__((__vector_size__(16))) int v4i;
typedef __attribute__((address_space(1))) v4i g_v4i;   // global (__device__)
typedef __attribute__((address_space(3))) v4i l_v4i;   // LDS (__shared__)
#endif

__device__ __forceinline__ void cp16_g2s(u16* dst, const u16* src) {
#ifdef HAVE_ASYNC_LDS
    __builtin_amdgcn_global_load_async_to_lds_b128(
        (g_v4i*)src, (l_v4i*)dst, 0, 0);
#else
    *(uint4*)dst = *(const uint4*)src;
#endif
}

__device__ __forceinline__ void wait_stage(bool has_next) {
#ifdef HAVE_ASYNC_LDS
#if __has_builtin(__builtin_amdgcn_s_wait_asynccnt)
    if (has_next) __builtin_amdgcn_s_wait_asynccnt(2);
    else          __builtin_amdgcn_s_wait_asynccnt(0);
#else
    if (has_next) asm volatile("s_wait_asynccnt 0x2" ::: "memory");
    else          asm volatile("s_wait_asynccnt 0x0" ::: "memory");
#endif
#endif
}

// ---------------------------------------------------------------------------
// WMMA bf16 GEMM: C = A @ B + bias.  A: bf16, up to 3 row-major K-concat
// sources (splits multiples of 32, selected uniformly per K-tile).  B: bf16.
// Output: f32 (optionally N-split across two destinations) or bf16 (Cb).
// Block tile 64x64, 8 waves, 2 WMMAs/wave/K-step, double-buffered LDS with
// async global->LDS copies (1x b128 per lane per tile).
// ---------------------------------------------------------------------------
__global__ __launch_bounds__(256) void gemm_wmma_bf16(
    const u16* __restrict__ A0, const u16* __restrict__ A1,
    const u16* __restrict__ A2, int ka1, int ka2,
    int lda0, int lda1, int lda2,
    const u16* __restrict__ Bw, int ldb,
    const float* __restrict__ bias0, const float* __restrict__ bias1, int nsplit,
    float* __restrict__ C0, float* __restrict__ C1, int ldc0, int ldc1,
    u16* __restrict__ Cb,
    int M, int Ntot, int K)
{
    __shared__ alignas(16) u16 As[2][BM][40];   // 32 data + pad; rows 80B (16B-aligned)
    __shared__ alignas(16) u16 Bs[2][BK][72];   // 64 data + pad; rows 144B (16B-aligned)

    const int m0   = blockIdx.x * BM;
    const int n0   = blockIdx.y * BN;
    const int tid  = threadIdx.x;
    const int wave = tid >> 5;
    const int lane = tid & 31;
    const int mi   = wave & 3;    // 4 m-subtiles of 16
    const int ni   = wave >> 2;   // 2 n-subtiles of 32 (2 wmmas each)

    // staging coordinates (one b128 per thread for A and for B)
    const int ar = tid >> 2, aseg = tid & 3;   // A: 64 rows x 4 segs of 8 bf16
    const int br = tid >> 3, bseg = tid & 7;   // B: 32 rows x 8 segs of 8 bf16

    v8f acc[2] = {{}, {}};

    // uniform per-K-tile A source select (splits are multiples of BK)
    auto stage = [&](int buf, int k0) {
        const u16* Ap; int lda, kloc;
        if      (k0 < ka1) { Ap = A0; lda = lda0; kloc = k0; }
        else if (k0 < ka2) { Ap = A1; lda = lda1; kloc = k0 - ka1; }
        else               { Ap = A2; lda = lda2; kloc = k0 - ka2; }
        cp16_g2s(&As[buf][ar][aseg * 8],
                 Ap + (size_t)(m0 + ar) * lda + kloc + aseg * 8);
        cp16_g2s(&Bs[buf][br][bseg * 8],
                 Bw + (size_t)(k0 + br) * ldb + n0 + bseg * 8);
    };

    stage(0, 0);
    int cur = 0;
    for (int k0 = 0; k0 < K; k0 += BK) {
        const bool has_next = (k0 + BK) < K;
        if (has_next) stage(cur ^ 1, k0 + BK);
        wait_stage(has_next);
        __syncthreads();

        // A fragment: lane m = mi*16+(lane&15); K halves selected by lane>>4
        union { v16bf v; unsigned int u[8]; } af;
        const int ml = mi * 16 + (lane & 15);
        const int kb = (lane >> 4) * 8;
        const unsigned int* arow = (const unsigned int*)&As[cur][ml][0];
        #pragma unroll
        for (int j = 0; j < 4; j++) af.u[j]     = arow[(kb >> 1) + j];
        #pragma unroll
        for (int j = 0; j < 4; j++) af.u[4 + j] = arow[8 + (kb >> 1) + j];

        const u16* bcol = &Bs[cur][0][lane & 15];
        #pragma unroll
        for (int nj = 0; nj < 2; nj++) {
            union { v16bf v; unsigned int u[8]; } bfr;
            const int nb = ni * 32 + nj * 16;
            #pragma unroll
            for (int j = 0; j < 8; j++) { const int kk = (j < 4) ? (kb + 2 * j) : (16 + kb + 2 * (j - 4)); bfr.u[j] = (unsigned int)bcol[kk * 72 + nb] | ((unsigned int)bcol[(kk + 1) * 72 + nb] << 16); }
            acc[nj] = __builtin_amdgcn_wmma_f32_16x16x32_bf16(
                          false, af.v, false, bfr.v, (short)0, acc[nj], false, false);
        }
        __syncthreads();
        cur ^= 1;
    }

    // store: M = m0 + mi*16 + 8*(lane>=16) + vgpr ; N = n0 + ni*32 + nj*16 + (lane&15)
    const int gm0 = m0 + mi * 16 + ((lane >> 4) * 8);
    #pragma unroll
    for (int nj = 0; nj < 2; nj++) {
        const int gn = n0 + ni * 32 + nj * 16 + (lane & 15);
        if (Cb) {
            float bv = bias0 ? bias0[gn] : 0.f;
            #pragma unroll
            for (int v = 0; v < 8; v++)
                Cb[(size_t)(gm0 + v) * ldc0 + gn] = f32_to_bf16(acc[nj][v] + bv);
        } else if (gn < nsplit) {
            float bv = bias0 ? bias0[gn] : 0.f;
            #pragma unroll
            for (int v = 0; v < 8; v++)
                C0[(size_t)(gm0 + v) * ldc0 + gn] = acc[nj][v] + bv;
        } else {
            float bv = bias1 ? bias1[gn - nsplit] : 0.f;
            #pragma unroll
            for (int v = 0; v < 8; v++)
                C1[(size_t)(gm0 + v) * ldc1 + (gn - nsplit)] = acc[nj][v] + bv;
        }
    }
}

// ---------------------------------------------------------------------------
// LayerNorm (fp32 in -> bf16 out). blockDim == C, one token per block.
// ---------------------------------------------------------------------------
__global__ void ln_kernel(const float* __restrict__ x, const float* __restrict__ w,
                          const float* __restrict__ b, u16* __restrict__ y, int C)
{
    __shared__ float red[128];
    size_t t = blockIdx.x;
    int c = threadIdx.x;
    float v = x[t * C + c];
    red[c] = v; __syncthreads();
    for (int s = C >> 1; s > 0; s >>= 1) { if (c < s) red[c] += red[c + s]; __syncthreads(); }
    float mu = red[0] / (float)C; __syncthreads();
    float d = v - mu;
    red[c] = d * d; __syncthreads();
    for (int s = C >> 1; s > 0; s >>= 1) { if (c < s) red[c] += red[c + s]; __syncthreads(); }
    float var = red[0] / (float)C;
    y[t * C + c] = f32_to_bf16(d * rsqrtf(var + 1e-6f) * w[c] + b[c]);
}

// fp32 -> bf16 weight conversion
__global__ void cvt_bf16_kernel(const float* __restrict__ src, u16* __restrict__ dst, int n)
{
    int i = blockIdx.x * blockDim.x + threadIdx.x;
    if (i < n) dst[i] = f32_to_bf16(src[i]);
}

// build combined [proj_w | proje_w] bf16 weight: 256 x 192
__global__ void build_wcat_kernel(const float* __restrict__ pw, const float* __restrict__ pew,
                                  u16* __restrict__ wcat)
{
    int i = blockIdx.x * blockDim.x + threadIdx.x;
    if (i >= 256 * 192) return;
    int k = i / 192, n = i - k * 192;
    float v = (n < 128) ? pw[k * 128 + n] : pew[k * 64 + (n - 128)];
    wcat[i] = f32_to_bf16(v);
}

// ---------------------------------------------------------------------------
// Depthwise 7x7 conv, pad 3 (bf16 in -> bf16 out). grid(H*W, B), blockDim=C.
// ---------------------------------------------------------------------------
__global__ void dwconv7_kernel(const u16* __restrict__ in, int ldin,
                               const float* __restrict__ w, const float* __restrict__ bias,
                               u16* __restrict__ out, int ldout, int H, int W)
{
    int p = blockIdx.x, b = blockIdx.y, c = threadIdx.x;
    int y = p / W, x = p - y * W;
    float s = bias ? bias[c] : 0.f;
    #pragma unroll
    for (int ky = 0; ky < 7; ky++) {
        int iy = y + ky - 3; if (iy < 0 || iy >= H) continue;
        #pragma unroll
        for (int kx = 0; kx < 7; kx++) {
            int ix = x + kx - 3; if (ix < 0 || ix >= W) continue;
            s += bf16_to_f32(in[(((size_t)b * H + iy) * W + ix) * ldin + c])
                 * w[c * 49 + ky * 7 + kx];
        }
    }
    out[(((size_t)b * H + y) * W + x) * ldout + c] = f32_to_bf16(s);
}

// local_rr = rr1 * rr3   (f32 x f32 -> bf16)
__global__ void lrr_kernel(const float* __restrict__ a, const float* __restrict__ b,
                           u16* __restrict__ o, long n)
{
    long i = (long)blockIdx.x * blockDim.x + threadIdx.x;
    if (i < n) o[i] = f32_to_bf16(a[i] * b[i]);
}

// prod = rgb*t ; diff = |rgb-t|   (f32 -> bf16)
__global__ void proddiff_kernel(const float* __restrict__ r, const float* __restrict__ t,
                                u16* __restrict__ p, u16* __restrict__ d, long n)
{
    long i = (long)blockIdx.x * blockDim.x + threadIdx.x;
    if (i < n) {
        float a = r[i], b = t[i];
        p[i] = f32_to_bf16(a * b);
        d[i] = f32_to_bf16(fabsf(a - b));
    }
}

__global__ void zero_kernel(float* __restrict__ p, int n)
{
    int i = blockIdx.x * blockDim.x + threadIdx.x;
    if (i < n) p[i] = 0.f;
}

// Per-batch stats for cosine channel attention. grid(chunks, B), 128 thr.
__global__ void stats_kernel(const u16* __restrict__ codi,
                             float* __restrict__ dot, float* __restrict__ n1,
                             float* __restrict__ n2, int HW, int ppb)
{
    __shared__ float red[128];
    int b = blockIdx.y, c = threadIdx.x;
    int p0 = blockIdx.x * ppb;
    float sdot = 0.f, sn2 = 0.f, sn1 = 0.f;
    for (int i = 0; i < ppb; i++) {
        int p = p0 + i; if (p >= HW) break;
        float v = bf16_to_f32(codi[((size_t)b * HW + p) * 128 + c]);
        red[c] = v; __syncthreads();
        for (int s = 64; s > 0; s >>= 1) { if (c < s) red[c] += red[c + s]; __syncthreads(); }
        float amap = red[0] * (1.f / 128.f);
        __syncthreads();
        sdot += amap * v; sn2 += v * v;
        if (c == 0) sn1 += amap * amap;
    }
    atomicAdd(&dot[b * 128 + c], sdot);
    atomicAdd(&n2[b * 128 + c], sn2);
    if (c == 0) atomicAdd(&n1[b], sn1);
}

// cos -> fc1 -> gelu(exact) -> fc2 -> sigmoid. grid(B), 128 thr.
__global__ void attc_kernel(const float* __restrict__ dot, const float* __restrict__ n1,
                            const float* __restrict__ n2, const float* __restrict__ fc1,
                            const float* __restrict__ fc2, float* __restrict__ attc)
{
    __shared__ float cosv[128];
    __shared__ float h[16];
    int b = blockIdx.x, c = threadIdx.x;
    float nn1 = sqrtf(n1[b]);
    float nn2 = sqrtf(n2[b * 128 + c]);
    cosv[c] = dot[b * 128 + c] / (nn1 * nn2 + 1e-6f);
    __syncthreads();
    if (c < 16) {
        float s = 0.f;
        for (int k = 0; k < 128; k++) s += cosv[k] * fc1[k * 16 + c];
        h[c] = 0.5f * s * (1.f + erff(s * 0.70710678118654752f));
    }
    __syncthreads();
    float s = 0.f;
    for (int j = 0; j < 16; j++) s += h[j] * fc2[j * 128 + c];
    attc[b * 128 + c] = 1.f / (1.f + __expf(-s));
}

// codi *= attc (broadcast over pixels), bf16 in-place
__global__ void scale_codi_kernel(u16* __restrict__ codi, const float* __restrict__ attc,
                                  int HW, long total)
{
    long i = (long)blockIdx.x * blockDim.x + threadIdx.x;
    if (i >= total) return;
    int c = (int)(i & 127);
    long pix = i >> 7;
    int b = (int)(pix / HW);
    codi[i] = f32_to_bf16(bf16_to_f32(codi[i]) * attc[b * 128 + c]);
}

// Adaptive 7x7 mean pool of [xn | xen] (192 ch, bf16 in). grid(49, B), 192 thr.
__global__ void pool_kernel(const u16* __restrict__ xn, const u16* __restrict__ xen,
                            float* __restrict__ pool, int H, int W)
{
    int b = blockIdx.y, widx = blockIdx.x, c = threadIdx.x;
    int wy = widx / 7, wx = widx - wy * 7;
    int ph = H / 7, pw = W / 7;
    float s = 0.f;
    for (int yy = 0; yy < ph; yy++)
        for (int xx = 0; xx < pw; xx++) {
            size_t p = ((size_t)b * H + (wy * ph + yy)) * W + (wx * pw + xx);
            s += (c < 128) ? bf16_to_f32(xn[p * 128 + c])
                           : bf16_to_f32(xen[p * 64 + (c - 128)]);
        }
    pool[((size_t)b * 49 + widx) * 192 + c] = s / (float)(ph * pw);
}

// Tiny q projection. grid(49, B), 64 thr.
__global__ void qproj_kernel(const float* __restrict__ pool, const float* __restrict__ qw,
                             const float* __restrict__ qb, float* __restrict__ q)
{
    int b = blockIdx.y, qi = blockIdx.x, n = threadIdx.x;
    const float* pr = &pool[((size_t)b * 49 + qi) * 192];
    float s = qb[n];
    for (int k = 0; k < 192; k++) s += pr[k] * qw[k * 64 + n];
    int nh = n >> 3, d = n & 7;
    q[(((size_t)b * 8 + nh) * 49 + qi) * 8 + d] = s;
}

// Online-softmax attention: one block per (b,nh,qi) row over HW keys.
__global__ __launch_bounds__(256) void attn_kernel(const float* __restrict__ q,
                                                   const float* __restrict__ kv,
                                                   float* __restrict__ g, int HW)
{
    __shared__ float sm[256], sl[256], sacc[256][8];
    int row = blockIdx.x;            // (b*8+nh)*49+qi
    int qi = row % 49;
    int bh = row / 49;
    int nh = bh & 7, b = bh >> 3;
    float qv[8];
    #pragma unroll
    for (int d = 0; d < 8; d++) qv[d] = q[(size_t)row * 8 + d];
    const float scale = 0.3535533905932738f;   // 8^-0.5

    float m = -1e30f, l = 0.f, acc[8];
    #pragma unroll
    for (int d = 0; d < 8; d++) acc[d] = 0.f;

    for (int j = threadIdx.x; j < HW; j += 256) {
        const float* kvp = &kv[((size_t)b * HW + j) * 128];
        float s = 0.f;
        #pragma unroll
        for (int d = 0; d < 8; d++) s += qv[d] * kvp[nh * 8 + d];
        s *= scale;
        float nm = fmaxf(m, s);
        float corr = __expf(m - nm), p = __expf(s - nm);
        l = l * corr + p;
        #pragma unroll
        for (int d = 0; d < 8; d++) acc[d] = acc[d] * corr + p * kvp[64 + nh * 8 + d];
        m = nm;
    }
    int t = threadIdx.x;
    sm[t] = m; sl[t] = l;
    #pragma unroll
    for (int d = 0; d < 8; d++) sacc[t][d] = acc[d];
    __syncthreads();
    for (int s2 = 128; s2 > 0; s2 >>= 1) {
        if (t < s2) {
            float m2 = sm[t + s2], nm = fmaxf(sm[t], m2);
            float c1 = __expf(sm[t] - nm), c2 = __expf(m2 - nm);
            sl[t] = sl[t] * c1 + sl[t + s2] * c2;
            #pragma unroll
            for (int d = 0; d < 8; d++)
                sacc[t][d] = sacc[t][d] * c1 + sacc[t + s2][d] * c2;
            sm[t] = nm;
        }
        __syncthreads();
    }
    if (t < 8)
        g[(((size_t)b * 64) + (nh * 8 + t)) * 49 + qi] = sacc[0][t] / sl[0];
}

// Bilinear 7x7 -> HxW (half-pixel, edge clamp), f32 in -> bf16 out.
__global__ void resize_kernel(const float* __restrict__ g, u16* __restrict__ out,
                              int H, int W)
{
    int p = blockIdx.x, b = blockIdx.y, c = threadIdx.x;
    int y = p / W, x = p - y * W;
    float sy = (y + 0.5f) * (7.f / (float)H) - 0.5f;
    float sx = (x + 0.5f) * (7.f / (float)W) - 0.5f;
    int y0 = (int)floorf(sy), x0 = (int)floorf(sx);
    float fy = sy - (float)y0, fx = sx - (float)x0;
    int y0c = min(max(y0, 0), 6), y1c = min(max(y0 + 1, 0), 6);
    int x0c = min(max(x0, 0), 6), x1c = min(max(x0 + 1, 0), 6);
    const float* gb = &g[((size_t)b * 64 + c) * 49];
    float v = gb[y0c * 7 + x0c] * (1.f - fy) * (1.f - fx)
            + gb[y0c * 7 + x1c] * (1.f - fy) * fx
            + gb[y1c * 7 + x0c] * fy * (1.f - fx)
            + gb[y1c * 7 + x1c] * fy * fx;
    out[((size_t)b * H * W + p) * 64 + c] = f32_to_bf16(v);
}

// ---------------------------------------------------------------------------
// Host-side plumbing
// ---------------------------------------------------------------------------
static void launch_gemm(hipStream_t st,
                        const u16* A0, const u16* A1, const u16* A2,
                        int ka1, int ka2, int lda0, int lda1, int lda2,
                        const u16* Bw, int ldb,
                        const float* bias0, const float* bias1, int nsplit,
                        float* C0, float* C1, int ldc0, int ldc1, u16* Cb,
                        int M, int Ntot, int K)
{
    dim3 grid(M / BM, Ntot / BN);
    gemm_wmma_bf16<<<grid, 256, 0, st>>>(A0, A1, A2, ka1, ka2, lda0, lda1, lda2,
                                         Bw, ldb, bias0, bias1, nsplit,
                                         C0, C1, ldc0, ldc1, Cb, M, Ntot, K);
}

extern "C" void kernel_launch(void* const* d_in, const int* in_sizes, int n_in,
                              void* d_out, int out_size, void* d_ws, size_t ws_size,
                              hipStream_t stream)
{
    const int B = 8, H = 112, W = 112;
    const int HW = H * W;
    const long N = (long)B * HW;                 // 100352 tokens

    const float* x       = (const float*)d_in[0];
    const float* x_e     = (const float*)d_in[1];
    const float* norm_w  = (const float*)d_in[2];
    const float* norm_b  = (const float*)d_in[3];
    const float* norme_w = (const float*)d_in[4];
    const float* norme_b = (const float*)d_in[5];
    const float* rr1_w   = (const float*)d_in[6];
    const float* rr1_b   = (const float*)d_in[7];
    const float* rr2_w   = (const float*)d_in[8];
    const float* rr2_b   = (const float*)d_in[9];
    const float* rr3_w   = (const float*)d_in[10];
    const float* rr3_b   = (const float*)d_in[11];
    const float* conv_w  = (const float*)d_in[12];
    const float* conv_b  = (const float*)d_in[13];
    const float* la1_w   = (const float*)d_in[14];
    const float* la1_b   = (const float*)d_in[15];
    const float* la2_w   = (const float*)d_in[16];
    const float* la2_b   = (const float*)d_in[17];
    const float* lac1_w  = (const float*)d_in[18];
    const float* lac1_b  = (const float*)d_in[19];
    const float* lac2_w  = (const float*)d_in[20];
    const float* lac2_b  = (const float*)d_in[21];
    const float* fc1_w   = (const float*)d_in[22];
    const float* fc2_w   = (const float*)d_in[23];
    const float* la3_w   = (const float*)d_in[24];
    const float* la3_b   = (const float*)d_in[25];
    const float* kv_w    = (const float*)d_in[26];
    const float* kv_b    = (const float*)d_in[27];
    const float* q_w     = (const float*)d_in[28];
    const float* q_b     = (const float*)d_in[29];
    const float* proj_w  = (const float*)d_in[30];
    const float* proj_b  = (const float*)d_in[31];
    const float* proje_w = (const float*)d_in[32];
    const float* proje_b = (const float*)d_in[33];
    (void)in_sizes; (void)n_in; (void)out_size; (void)ws_size;

    // ---- workspace layout: bf16 region then f32 region ----
    u16* wsb   = (u16*)d_ws;
    u16* xnb   = wsb;                 // N*128
    u16* xenb  = xnb   + N * 128;     // N*64
    u16* rr2b  = xenb  + N * 64;      // N*128
    u16* rr2cb = rr2b  + N * 128;     // N*128
    u16* lrrb  = rr2cb + N * 128;     // N*128
    u16* prodb = lrrb  + N * 128;     // N*64
    u16* diffb = prodb + N * 64;      // N*64
    u16* codib = diffb + N * 64;      // N*128
    u16* lrxb  = codib + N * 128;     // N*64
    u16* gresb = lrxb  + N * 64;      // N*64
    u16* wbrr1 = gresb + N * 64;      // 16384
    u16* wbrr2 = wbrr1 + 16384;
    u16* wbrr3 = wbrr2 + 16384;
    u16* wbkv  = wbrr3 + 16384;
    u16* wbla1 = wbkv  + 16384;       // 8192
    u16* wbla2 = wbla1 + 8192;        // 4096
    u16* wbla3 = wbla2 + 4096;        // 8192
    u16* wcat  = wbla3 + 8192;        // 256*192 = 49152

    float* wsf  = (float*)(wcat + 49152);
    float* bufA = wsf;                // N*128
    float* bufB = bufA + N * 128;     // N*128
    float* dot  = bufB + N * 128;     // 1024
    float* n1   = dot  + 1024;        // 8
    float* n2   = n1   + 8;           // 1024
    float* attc = n2   + 1024;        // 1024
    float* pool = attc + 1024;        // 8*49*192
    float* qbuf = pool + 8 * 49 * 192;
    float* gsm  = qbuf + 8 * 8 * 49 * 8;

    float* rgb  = bufA;               // N*64 (after rr1 consumed)
    float* tbr  = bufA + N * 64;      // N*64

    float* out   = (float*)d_out;     // N*128
    float* out_e = out + N * 128;     // N*64

    // 0) weights -> bf16 (once per call; tiny)
    cvt_bf16_kernel<<<64, 256, 0, stream>>>(rr1_w, wbrr1, 16384);
    cvt_bf16_kernel<<<64, 256, 0, stream>>>(rr2_w, wbrr2, 16384);
    cvt_bf16_kernel<<<64, 256, 0, stream>>>(rr3_w, wbrr3, 16384);
    cvt_bf16_kernel<<<64, 256, 0, stream>>>(kv_w,  wbkv,  16384);
    cvt_bf16_kernel<<<32, 256, 0, stream>>>(la1_w, wbla1, 8192);
    cvt_bf16_kernel<<<16, 256, 0, stream>>>(la2_w, wbla2, 4096);
    cvt_bf16_kernel<<<32, 256, 0, stream>>>(la3_w, wbla3, 8192);
    build_wcat_kernel<<<192, 256, 0, stream>>>(proj_w, proje_w, wcat);

    // 1) LayerNorms -> bf16
    ln_kernel<<<(unsigned)N, 128, 0, stream>>>(x,   norm_w,  norm_b,  xnb,  128);
    ln_kernel<<<(unsigned)N,  64, 0, stream>>>(x_e, norme_w, norme_b, xenb,  64);

    // 2) local_rr branch
    launch_gemm(stream, xnb, nullptr, nullptr, 128, 128, 128, 0, 0,
                wbrr1, 128, rr1_b, nullptr, 1 << 30,
                bufA, nullptr, 128, 0, nullptr, (int)N, 128, 128);
    launch_gemm(stream, xnb, nullptr, nullptr, 128, 128, 128, 0, 0,
                wbrr2, 128, rr2_b, nullptr, 1 << 30,
                nullptr, nullptr, 128, 0, rr2b, (int)N, 128, 128);
    dwconv7_kernel<<<dim3(HW, B), 128, 0, stream>>>(rr2b, 128, conv_w, conv_b,
                                                    rr2cb, 128, H, W);
    launch_gemm(stream, rr2cb, nullptr, nullptr, 128, 128, 128, 0, 0,
                wbrr3, 128, rr3_b, nullptr, 1 << 30,
                bufB, nullptr, 128, 0, nullptr, (int)N, 128, 128);
    {
        long n = N * 128;
        lrr_kernel<<<(unsigned)((n + 255) / 256), 256, 0, stream>>>(bufA, bufB, lrrb, n);
    }

    // 3) LocalAttentionRGBT branch
    launch_gemm(stream, xnb, nullptr, nullptr, 128, 128, 128, 0, 0,
                wbla1, 64, la1_b, nullptr, 1 << 30,
                rgb, nullptr, 64, 0, nullptr, (int)N, 64, 128);
    launch_gemm(stream, xenb, nullptr, nullptr, 64, 64, 64, 0, 0,
                wbla2, 64, la2_b, nullptr, 1 << 30,
                tbr, nullptr, 64, 0, nullptr, (int)N, 64, 64);
    {
        long n = N * 64;
        proddiff_kernel<<<(unsigned)((n + 255) / 256), 256, 0, stream>>>(rgb, tbr,
                                                                         prodb, diffb, n);
    }
    dwconv7_kernel<<<dim3(HW, B), 64, 0, stream>>>(prodb, 64, lac1_w, lac1_b,
                                                   codib,      128, H, W);
    dwconv7_kernel<<<dim3(HW, B), 64, 0, stream>>>(diffb, 64, lac2_w, lac2_b,
                                                   codib + 64, 128, H, W);
    zero_kernel<<<(2056 + 255) / 256, 256, 0, stream>>>(dot, 1024 + 8 + 1024);
    stats_kernel<<<dim3(49, B), 128, 0, stream>>>(codib, dot, n1, n2, HW, 256);
    attc_kernel<<<B, 128, 0, stream>>>(dot, n1, n2, fc1_w, fc2_w, attc);
    {
        long n = N * 128;
        scale_codi_kernel<<<(unsigned)((n + 255) / 256), 256, 0, stream>>>(codib, attc, HW, n);
    }
    launch_gemm(stream, codib, nullptr, nullptr, 128, 128, 128, 0, 0,
                wbla3, 64, la3_b, nullptr, 1 << 30,
                nullptr, nullptr, 64, 0, lrxb, (int)N, 64, 128);

    // 4) global pooled attention
    launch_gemm(stream, xnb, nullptr, nullptr, 128, 128, 128, 0, 0,
                wbkv, 128, kv_b, nullptr, 1 << 30,
                bufB, nullptr, 128, 0, nullptr, (int)N, 128, 128);
    pool_kernel<<<dim3(49, B), 192, 0, stream>>>(xnb, xenb, pool, H, W);
    qproj_kernel<<<dim3(49, B), 64, 0, stream>>>(pool, q_w, q_b, qbuf);
    attn_kernel<<<B * 8 * 49, 256, 0, stream>>>(qbuf, bufB, gsm, HW);
    resize_kernel<<<dim3(HW, B), 64, 0, stream>>>(gsm, gresb, H, W);

    // 5) fused output projections: xc = [local_rr | g | local_rx] (K=256),
    //    cols [0,128) -> out (proj), cols [128,192) -> out_e (proje)
    launch_gemm(stream, lrrb, gresb, lrxb, 128, 192, 128, 64, 64,
                wcat, 192, proj_b, proje_b, 128,
                out, out_e, 128, 64, nullptr, (int)N, 192, 256);
}
